// LongformerEmbed_73916387164493
// MI455X (gfx1250) — matrix operand order repaired and checked by reading, not compiled, not run on hardware
//
#include <hip/hip_runtime.h>

#define DEVFN __device__ __forceinline__

typedef __bf16 bf16_t;
typedef __attribute__((ext_vector_type(16))) __bf16 bf16x16;
typedef __attribute__((ext_vector_type(8)))  __bf16 bf16x8;
typedef __attribute__((ext_vector_type(8)))  float  f32x8;

// longformer-base-4096 config
static constexpr int S  = 4096;
static constexpr int H  = 768;
static constexpr int NH = 12;
static constexpr int DH = 64;
static constexpr int FF = 3072;
static constexpr int L  = 12;
static constexpr int Wn = 256;   // window chunk
static constexpr int NC = 16;    // S / Wn
static constexpr int E  = 512;
static constexpr int H3 = 3 * H; // 2304 fused qkv width

// ---------------------------------------------------------------------------
// WMMA helpers (CDNA5 16x16x32 bf16, f32 accumulate)
// ---------------------------------------------------------------------------
DEVFN bf16x16 ld_frag(const bf16_t* p0, const bf16_t* p1) {
  bf16x8 lo = *reinterpret_cast<const bf16x8*>(p0);
  bf16x8 hi = *reinterpret_cast<const bf16x8*>(p1);
  return __builtin_shufflevector(lo, hi, 0,1,2,3,4,5,6,7,8,9,10,11,12,13,14,15);
}

DEVFN f32x8 wmma_bf16(bf16x16 a, bf16x16 b, f32x8 c) {
  return __builtin_amdgcn_wmma_f32_16x16x32_bf16(false, a, false, b, (short)0, c,
                                                 false, false);
}

// CDNA5 async global->LDS DMA (per-lane 16B), tracked by ASYNCcnt
DEVFN unsigned lds_off(const void* p) {
  return (unsigned)(unsigned long long)p;   // low 32 bits = LDS byte offset
}
DEVFN void async_cp16(unsigned lds, const bf16_t* g) {
  asm volatile("global_load_async_to_lds_b128 %0, %1, off"
               :: "v"(lds), "v"(g) : "memory");
}
DEVFN void wait_async0() { asm volatile("s_wait_asynccnt 0x0" ::: "memory"); }

// ---------------------------------------------------------------------------
// Weight conversion: f32 [K,N] row-major -> bf16 [N,K] (transposed for B-frags)
// ---------------------------------------------------------------------------
__global__ void k_transpose_bf16(const float* __restrict__ in,
                                 bf16_t* __restrict__ out, int K, int N) {
  int idx = blockIdx.x * 256 + threadIdx.x;      // over N*K, exact multiple
  int n = idx / K;
  int k = idx - n * K;
  out[idx] = (bf16_t)in[(size_t)k * N + n];
}

__global__ void k_fuse_qkv_bias(const float* __restrict__ bq,
                                const float* __restrict__ bk,
                                const float* __restrict__ bv,
                                float* __restrict__ out) {
  int i = blockIdx.x * 256 + threadIdx.x;        // 2304
  float v;
  if (i < H)            v = bq[i];
  else if (i < 2 * H)   v = bk[i - H];
  else                  v = bv[i - 2 * H];
  out[i] = v;
}

// vT[(h*64+d)][t] = qkv[t][1536 + h*64 + d]
__global__ void k_transpose_v(const bf16_t* __restrict__ qkv,
                              bf16_t* __restrict__ vT) {
  int idx = blockIdx.x * 256 + threadIdx.x;      // over H*S
  int hd = idx >> 12;                            // / 4096
  int t  = idx & (S - 1);
  vT[idx] = qkv[(size_t)t * H3 + 2 * H + hd];
}

// ---------------------------------------------------------------------------
// GEMM: C[M,N] = A[M,K] (bf16, row-major) x BT[N,K] (bf16) + bias
// MODE 0: bf16 out ; MODE 1: bf16 out with exact GELU ; MODE 2: f32 out
// 512 thr = 16 waves (4m x 4n), wave tile 32x64, block tile 128x256.
// A/B tiles staged via async global->LDS DMA, double buffered.
// ---------------------------------------------------------------------------
template <int MODE>
__global__ __launch_bounds__(512) void k_gemm(const bf16_t* __restrict__ A,
                                              const bf16_t* __restrict__ BT,
                                              const float* __restrict__ bias,
                                              void* __restrict__ Out,
                                              int M, int N, int K) {
  __shared__ __align__(16) bf16_t Al[2][128][32];   // 16 KB
  __shared__ __align__(16) bf16_t Bl[2][256][32];   // 32 KB

  const int tid     = threadIdx.x;
  const int wave    = tid >> 5;
  const int lane    = tid & 31;
  const int lane_m  = lane & 15;
  const int lane_hi = lane >> 4;
  const int mw = wave & 3, nw = wave >> 2;
  const int mblk = blockIdx.y * 128;
  const int nblk = blockIdx.x * 256;
  const int m_base = mblk + mw * 32;
  const int n_base = nblk + nw * 64;

  // staging coordinates: A = 128 rows x 4 16B-chunks (512), B = 256 x 4 (1024)
  const int ar  = tid >> 2, ac = (tid & 3) * 8;
  const int bc0 = tid >> 1, bk0 = (tid & 1) * 16;   // 2 chunks -> split pairs
  // unused (bc1 = bc0 + 256); the second B chunk is row bc0 at k bk0 + 8
  const int idx1 = tid + 512;
  const int bc1 = idx1 >> 1, bk1 = (idx1 & 1) * 16;

  auto stage = [&](int buf, int k0) {
    async_cp16(lds_off(&Al[buf][ar][ac]),
               A + (size_t)(mblk + ar) * K + k0 + ac);
    async_cp16(lds_off(&Bl[buf][bc0][bk0]),
               BT + (size_t)(nblk + bc0) * K + k0 + bk0);
    async_cp16(lds_off(&Bl[buf][bc0][bk0 + 8]),
               BT + (size_t)(nblk + bc0) * K + k0 + bk0 + 8);
    (void)bc1; (void)bk1;
  };

  f32x8 acc[2][4];
  for (int i = 0; i < 2; ++i)
    for (int j = 0; j < 4; ++j)
      acc[i][j] = f32x8{0.f, 0.f, 0.f, 0.f, 0.f, 0.f, 0.f, 0.f};

  const int nk = K >> 5;
  stage(0, 0);
  for (int ks = 0; ks < nk; ++ks) {
    wait_async0();
    __syncthreads();
    if (ks + 1 < nk) stage((ks + 1) & 1, (ks + 1) * 32);
    const int buf = ks & 1;

    bf16x16 af[2], bfr[4];
#pragma unroll
    for (int tm = 0; tm < 2; ++tm) {
      const bf16_t* ap = &Al[buf][mw * 32 + tm * 16 + lane_m][lane_hi * 8];
      af[tm] = ld_frag(ap, ap + 16);
    }
#pragma unroll
    for (int tn = 0; tn < 4; ++tn) {
      const bf16_t* bp = &Bl[buf][nw * 64 + tn * 16 + lane_m][lane_hi * 8];
      bfr[tn] = ld_frag(bp, bp + 16);
    }
#pragma unroll
    for (int tm = 0; tm < 2; ++tm)
#pragma unroll
      for (int tn = 0; tn < 4; ++tn)
        acc[tm][tn] = wmma_bf16(af[tm], bfr[tn], acc[tm][tn]);
    __syncthreads();
  }

  for (int tm = 0; tm < 2; ++tm)
    for (int tn = 0; tn < 4; ++tn) {
      int col = n_base + tn * 16 + lane_m;
      float bv = bias[col];
      for (int r = 0; r < 8; ++r) {
        int row = m_base + tm * 16 + lane_hi * 8 + r;
        float v = acc[tm][tn][r] + bv;
        if (MODE == 1) v = 0.5f * v * (1.0f + erff(v * 0.70710678118654752f));
        if (MODE == 2)
          ((float*)Out)[(size_t)row * N + col] = v;
        else
          ((bf16_t*)Out)[(size_t)row * N + col] = (bf16_t)v;
      }
    }
}

// ---------------------------------------------------------------------------
// Sliding-window attention, flash-style. One block per (head, chunk).
// 512 threads = 16 waves; wave w handles query rows [w*16, w*16+16).
// qkv: [S, 2304] bf16 (q | k | v). vT: [H, S] bf16 (per head,dh rows).
// out: attn [S, H] bf16.
// ---------------------------------------------------------------------------
__global__ __launch_bounds__(512) void k_attn(const bf16_t* __restrict__ qkv,
                                              const bf16_t* __restrict__ vT,
                                              const int* __restrict__ amask,
                                              bf16_t* __restrict__ attn) {
  const int hd_id = blockIdx.x / NC;   // head
  const int c     = blockIdx.x % NC;   // chunk
  const int wave    = threadIdx.x >> 5;
  const int lane    = threadIdx.x & 31;
  const int lane_m  = lane & 15;
  const int lane_hi = lane >> 4;
  const int i0 = wave * 16;            // first query row of this wave

  __shared__ __align__(16) bf16_t Pl[16][16][32];  // per-wave P tile (16KB)

  // Preload Q fragments (2 contraction steps over DH=64)
  bf16x16 qf[2];
  {
    int row = i0 + lane_m;
    int t   = c * Wn + row;
    const bf16_t* qb = qkv + (size_t)t * H3 + hd_id * DH;
    for (int ks = 0; ks < 2; ++ks) {
      const bf16_t* qp = qb + ks * 32 + lane_hi * 8;
      qf[ks] = ld_frag(qp, qp + 16);
    }
  }

  float mrun[8], lrun[8];
  for (int r = 0; r < 8; ++r) { mrun[r] = -1e30f; lrun[r] = 0.f; }
  f32x8 ctx[4];
  for (int tn = 0; tn < 4; ++tn)
    ctx[tn] = f32x8{0.f, 0.f, 0.f, 0.f, 0.f, 0.f, 0.f, 0.f};

  for (int jb = 0; jb < 24; ++jb) {      // 24 blocks of 32 keys = 3W
    const int j0 = jb * 32;

    // ---- scores: S[16 x 32] = Q (16x64) x K^T ----
    f32x8 s[2];
#pragma unroll
    for (int tj = 0; tj < 2; ++tj) {
      int jcol = j0 + tj * 16 + lane_m;
      int kt   = c * Wn + jcol - Wn;
      int ktc  = kt < 0 ? 0 : (kt > S - 1 ? S - 1 : kt);
      const bf16_t* kb = qkv + (size_t)ktc * H3 + H + hd_id * DH;
      f32x8 st = f32x8{0.f, 0.f, 0.f, 0.f, 0.f, 0.f, 0.f, 0.f};
#pragma unroll
      for (int ks = 0; ks < 2; ++ks) {
        int dh0 = ks * 32 + lane_hi * 8;
        bf16x16 kf = ld_frag(kb + dh0, kb + dh0 + 16);
        st = wmma_bf16(qf[ks], kf, st);
      }
      s[tj] = st;
    }

    // ---- mask + scale (branchless: clamped unconditional mask load) ----
#pragma unroll
    for (int tj = 0; tj < 2; ++tj) {
      int jcol = j0 + tj * 16 + lane_m;
      int kt   = c * Wn + jcol - Wn;
      int ktc  = kt < 0 ? 0 : (kt > S - 1 ? S - 1 : kt);
      int mval = amask[ktc];                       // safe, clamped index
      int kvalid = (int)(kt >= 0) & (int)(kt < S) & (int)(mval != 0);
#pragma unroll
      for (int r = 0; r < 8; ++r) {
        int qi = i0 + lane_hi * 8 + r;
        int ok = kvalid & (int)(jcol >= qi) & (int)(jcol <= qi + 2 * Wn);
        float sv = s[tj][r] * 0.125f;     // 1/sqrt(64)
        s[tj][r] = ok ? sv : -1e9f;
      }
    }

    // ---- online softmax update ----
#pragma unroll
    for (int r = 0; r < 8; ++r) {
      float smax = fmaxf(s[0][r], s[1][r]);
      for (int off = 1; off < 16; off <<= 1)
        smax = fmaxf(smax, __shfl_xor(smax, off, 16));
      float mnew  = fmaxf(mrun[r], smax);
      float scale = __expf(mrun[r] - mnew);
      float p0 = __expf(s[0][r] - mnew);
      float p1 = __expf(s[1][r] - mnew);
      float psum = p0 + p1;
      for (int off = 1; off < 16; off <<= 1)
        psum += __shfl_xor(psum, off, 16);
      lrun[r] = lrun[r] * scale + psum;
      mrun[r] = mnew;
      for (int tn = 0; tn < 4; ++tn) ctx[tn][r] *= scale;
      int prow = lane_hi * 8 + r;
      Pl[wave][prow][lane_m]      = (bf16_t)p0;
      Pl[wave][prow][16 + lane_m] = (bf16_t)p1;
    }

    // ---- ctx += P (16x32) x V (32x64) ----
    const bf16_t* pr = &Pl[wave][lane_m][lane_hi * 16];
    bf16x16 pf = ld_frag(pr, pr + 8);
    int kt0 = c * Wn + j0 + lane_hi * 16 - Wn;
    int a0  = kt0     < 0 ? 0 : (kt0     > S - 8 ? S - 8 : kt0);
    int a1  = kt0 + 8 < 0 ? 0 : (kt0 + 8 > S - 8 ? S - 8 : kt0 + 8);
#pragma unroll
    for (int tn = 0; tn < 4; ++tn) {
      int dh = tn * 16 + lane_m;
      const bf16_t* vp = vT + (size_t)(hd_id * DH + dh) * S;
      bf16x16 vf = ld_frag(vp + a0, vp + a1);
      ctx[tn] = wmma_bf16(pf, vf, ctx[tn]);
    }
  }

  // ---- write out: attn[t, h*64 + dh] = ctx / l ----
  for (int tn = 0; tn < 4; ++tn) {
    int dh = tn * 16 + lane_m;
    for (int r = 0; r < 8; ++r) {
      int row = i0 + lane_hi * 8 + r;
      int t   = c * Wn + row;
      attn[(size_t)t * H + hd_id * DH + dh] = (bf16_t)(ctx[tn][r] / lrun[r]);
    }
  }
}

// ---------------------------------------------------------------------------
// Embedding gather + LayerNorm (one block per token, 256 threads, 3 elems ea)
// ---------------------------------------------------------------------------
DEVFN float block_sum(float v, float* red) {
  int t = threadIdx.x;
  red[t] = v;
  __syncthreads();
  for (int s = 128; s > 0; s >>= 1) {
    if (t < s) red[t] += red[t + s];
    __syncthreads();
  }
  float r = red[0];
  __syncthreads();
  return r;
}

__global__ __launch_bounds__(256) void k_embed_ln(
    const int* __restrict__ ids, const float* __restrict__ wemb,
    const float* __restrict__ pemb, const float* __restrict__ temb,
    const float* __restrict__ g, const float* __restrict__ b,
    float* __restrict__ h, bf16_t* __restrict__ hb) {
  __shared__ float red[256];
  int t  = blockIdx.x;
  int id = ids[t];
  float x[3], sum = 0.f;
  for (int e = 0; e < 3; ++e) {
    int i = threadIdx.x + e * 256;
    x[e] = wemb[(size_t)id * H + i] + pemb[(size_t)t * H + i] + temb[i];
    sum += x[e];
  }
  float mu = block_sum(sum, red) * (1.0f / H);
  float s2 = 0.f;
  for (int e = 0; e < 3; ++e) { float d = x[e] - mu; s2 += d * d; }
  float var = block_sum(s2, red) * (1.0f / H);
  float rstd = rsqrtf(var + 1e-5f);
  for (int e = 0; e < 3; ++e) {
    int i = threadIdx.x + e * 256;
    float y = (x[e] - mu) * rstd * g[i] + b[i];
    h[(size_t)t * H + i]  = y;
    hb[(size_t)t * H + i] = (bf16_t)y;
  }
}

// h = LN(h + a), also emit bf16 copy
__global__ __launch_bounds__(256) void k_add_ln(
    float* __restrict__ h, const float* __restrict__ a,
    const float* __restrict__ g, const float* __restrict__ b,
    bf16_t* __restrict__ hb) {
  __shared__ float red[256];
  int t = blockIdx.x;
  float x[3], sum = 0.f;
  for (int e = 0; e < 3; ++e) {
    int i = threadIdx.x + e * 256;
    x[e] = h[(size_t)t * H + i] + a[(size_t)t * H + i];
    sum += x[e];
  }
  float mu = block_sum(sum, red) * (1.0f / H);
  float s2 = 0.f;
  for (int e = 0; e < 3; ++e) { float d = x[e] - mu; s2 += d * d; }
  float var = block_sum(s2, red) * (1.0f / H);
  float rstd = rsqrtf(var + 1e-5f);
  for (int e = 0; e < 3; ++e) {
    int i = threadIdx.x + e * 256;
    float y = (x[e] - mu) * rstd * g[i] + b[i];
    h[(size_t)t * H + i]  = y;
    hb[(size_t)t * H + i] = (bf16_t)y;
  }
}

// ---------------------------------------------------------------------------
// CLS head: out = tanh(h[0] @ dense_w + db) @ out_w + ob   (tiny; f32 VALU)
// ---------------------------------------------------------------------------
__global__ __launch_bounds__(256) void k_head(const float* __restrict__ h0,
                                              const float* __restrict__ dw,
                                              const float* __restrict__ db,
                                              const float* __restrict__ ow,
                                              const float* __restrict__ ob,
                                              float* __restrict__ out) {
  __shared__ float cls[H];
  for (int j = threadIdx.x; j < H; j += 256) {
    float acc = db[j];
    for (int i = 0; i < H; ++i) acc += h0[i] * dw[(size_t)i * H + j];
    cls[j] = tanhf(acc);
  }
  __syncthreads();
  for (int e = threadIdx.x; e < E; e += 256) {
    float acc = ob[e];
    for (int i = 0; i < H; ++i) acc += cls[i] * ow[(size_t)i * E + e];
    out[e] = acc;
  }
}

// ---------------------------------------------------------------------------
// Host orchestration
// ---------------------------------------------------------------------------
extern "C" void kernel_launch(void* const* d_in, const int* in_sizes, int n_in,
                              void* d_out, int out_size, void* d_ws,
                              size_t ws_size, hipStream_t stream) {
  (void)in_sizes; (void)n_in; (void)out_size; (void)ws_size;
  const float* word_emb = (const float*)d_in[0];
  const float* pos_emb  = (const float*)d_in[1];
  const float* tok_emb  = (const float*)d_in[2];
  const float* eg       = (const float*)d_in[3];
  const float* eb       = (const float*)d_in[4];
  const float* wq = (const float*)d_in[5];
  const float* bq = (const float*)d_in[6];
  const float* wk = (const float*)d_in[7];
  const float* bk = (const float*)d_in[8];
  const float* wv = (const float*)d_in[9];
  const float* bv = (const float*)d_in[10];
  const float* wo = (const float*)d_in[11];
  const float* bo = (const float*)d_in[12];
  const float* l1g = (const float*)d_in[13];
  const float* l1b = (const float*)d_in[14];
  const float* w1 = (const float*)d_in[15];
  const float* b1 = (const float*)d_in[16];
  const float* w2 = (const float*)d_in[17];
  const float* b2 = (const float*)d_in[18];
  const float* l2g = (const float*)d_in[19];
  const float* l2b = (const float*)d_in[20];
  const float* dense_w = (const float*)d_in[21];
  const float* dense_b = (const float*)d_in[22];
  const float* out_w   = (const float*)d_in[23];
  const float* out_b   = (const float*)d_in[24];
  const int*   ids     = (const int*)d_in[25];
  const int*   amask   = (const int*)d_in[26];

  char* ws = (char*)d_ws;
  auto bump = [&](size_t bytes) {
    char* p = ws;
    ws += (bytes + 255) & ~(size_t)255;
    return p;
  };
  float*  h      = (float*)bump((size_t)S * H * 4);
  bf16_t* hb     = (bf16_t*)bump((size_t)S * H * 2);
  bf16_t* qkv    = (bf16_t*)bump((size_t)S * H3 * 2);
  bf16_t* vT     = (bf16_t*)bump((size_t)H * S * 2);
  bf16_t* attn   = (bf16_t*)bump((size_t)S * H * 2);
  float*  yf32   = (float*)bump((size_t)S * H * 4);
  bf16_t* ff1    = (bf16_t*)bump((size_t)S * FF * 2);
  bf16_t* WqkvT  = (bf16_t*)bump((size_t)H3 * H * 2);
  float*  bqkv   = (float*)bump((size_t)H3 * 4);
  bf16_t* WoT    = (bf16_t*)bump((size_t)H * H * 2);
  bf16_t* W1T    = (bf16_t*)bump((size_t)FF * H * 2);
  bf16_t* W2T    = (bf16_t*)bump((size_t)H * FF * 2);

  // Embedding + LN
  k_embed_ln<<<S, 256, 0, stream>>>(ids, word_emb, pos_emb, tok_emb, eg, eb, h, hb);

  const int grHH = (H * H) / 256;    // 2304 blocks
  const int grHF = (H * FF) / 256;   // 9216 blocks

  for (int l = 0; l < L; ++l) {
    const size_t oHH = (size_t)l * H * H;
    const size_t oHF = (size_t)l * H * FF;
    // Per-layer weight prep: bf16 transposed copies
    k_transpose_bf16<<<grHH, 256, 0, stream>>>(wq + oHH, WqkvT,             H, H);
    k_transpose_bf16<<<grHH, 256, 0, stream>>>(wk + oHH, WqkvT + (size_t)H * H,     H, H);
    k_transpose_bf16<<<grHH, 256, 0, stream>>>(wv + oHH, WqkvT + (size_t)2 * H * H, H, H);
    k_fuse_qkv_bias<<<H3 / 256, 256, 0, stream>>>(bq + (size_t)l * H, bk + (size_t)l * H,
                                                  bv + (size_t)l * H, bqkv);
    k_transpose_bf16<<<grHH, 256, 0, stream>>>(wo + oHH, WoT, H, H);
    k_transpose_bf16<<<grHF, 256, 0, stream>>>(w1 + oHF, W1T, H, FF);
    k_transpose_bf16<<<grHF, 256, 0, stream>>>(w2 + oHF, W2T, FF, H);

    // QKV projection (fused): [S,H] x [H,3H]
    k_gemm<0><<<dim3(H3 / 256, S / 128), 512, 0, stream>>>(hb, WqkvT, bqkv, qkv,
                                                           S, H3, H);
    // V transpose for attention B-frags
    k_transpose_v<<<(H * S) / 256, 256, 0, stream>>>(qkv, vT);
    // Sliding-window attention
    k_attn<<<NH * NC, 512, 0, stream>>>(qkv, vT, amask, attn);
    // Output projection -> f32
    k_gemm<2><<<dim3(H / 256, S / 128), 512, 0, stream>>>(attn, WoT,
                                                          bo + (size_t)l * H, yf32,
                                                          S, H, H);
    k_add_ln<<<S, 256, 0, stream>>>(h, yf32, l1g + (size_t)l * H,
                                    l1b + (size_t)l * H, hb);
    // FFN
    k_gemm<1><<<dim3(FF / 256, S / 128), 512, 0, stream>>>(hb, W1T,
                                                           b1 + (size_t)l * FF, ff1,
                                                           S, FF, H);
    k_gemm<2><<<dim3(H / 256, S / 128), 512, 0, stream>>>(ff1, W2T,
                                                          b2 + (size_t)l * H, yf32,
                                                          S, H, FF);
    k_add_ln<<<S, 256, 0, stream>>>(h, yf32, l2g + (size_t)l * H,
                                    l2b + (size_t)l * H, hb);
  }

  // CLS head on token 0
  k_head<<<1, 256, 0, stream>>>(h, dense_w, dense_b, out_w, out_b, (float*)d_out);
}
